// Self_attention_26714696581522
// MI455X (gfx1250) — compile-verified, matrix operand order checked by reading, not run on hardware
//
// CDNA5 / gfx1250 self-attention pipeline for MI455X (wave32, WMMA bf16).
//
//   K1: QK = h@Wqk + bqk, deinterleaved to Qbf/Kbf (bf16)        [wmma bf16]
//   K2: HW = h@weight (bf16)                                     [wmma bf16]
//   K3: at = 0.125 * Qbf @ Kbf^T (fp32 out)                      [wmma bf16, async-LDS dbl-buf]
//   K4: per-row rank-170 threshold (bitonic sort) + softmax -> attn (bf16)
//   K5: out = attn @ HW + bias (fp32)                            [wmma bf16, async-LDS dbl-buf]
//
// Workspace use: ~63 MB.

#include <hip/hip_runtime.h>

typedef __bf16 bf16_t;
typedef __attribute__((ext_vector_type(16))) __bf16 v16bf;
typedef __attribute__((ext_vector_type(8)))  __bf16 v8bf;
typedef __attribute__((ext_vector_type(8)))  float  v8f;

#define TILE_M 128
#define TILE_N 64
#define TILE_K 32
#define LDSS   40   // bf16 per LDS row: 32 data + 8 pad -> 16B aligned b128 reads, conflict-free

union Frag { v16bf v; v8bf h[2]; };

// ADT/BDT: 0 = fp32 global source (convert to bf16), 1 = bf16 global source (async copy)
// BT: B supplied as [Ncols][K] row-major (i.e. B^T); else [K][Ncols]
// MODE: 0 = +bias, even/odd column deinterleave to bf16 C0/C1 (ldc = Ncols/2)
//       1 = bf16 store
//       2 = fp32 store * scale
//       3 = fp32 store + bias
template<int ADT, int BDT, bool BT, int MODE>
__global__ __launch_bounds__(256)
void gemm_wmma(const void* __restrict__ A, int lda, long sA,
               const void* __restrict__ B, int ldb, long sB,
               void* __restrict__ C0, void* __restrict__ C1, int ldc, long sC,
               const float* __restrict__ bias, float scale, int K)
{
    __shared__ __align__(16) bf16_t As[2][TILE_M * LDSS];   // 2 x 10240 B
    __shared__ __align__(16) bf16_t Bs[2][TILE_N * LDSS];   // 2 x  5120 B (B^T tiles)

    const int tid  = threadIdx.x;
    const int lane = tid & 31;
    const int wv   = tid >> 5;        // 8 waves
    const int wm   = wv & 3;          // 4 m-waves * 32 rows = 128
    const int wn   = wv >> 2;         // 2 n-waves * 32 cols = 64
    const int fr   = lane & 15;       // fragment row/col within 16
    const int kh   = lane >> 4;       // K-half selector (bf16 A/B layout)

    const long aRow0 = (long)blockIdx.y * TILE_M;
    const long bCol0 = (long)blockIdx.x * TILE_N;

    // ---- tile staging (uniform trip counts, loads batched before converts) ----
    auto stage = [&](int buf, int kt) {
        bf16_t* as = &As[buf][0];
        bf16_t* bs = &Bs[buf][0];
        // A tile: TILE_M x TILE_K
        if constexpr (ADT == 0) {
            const float* Ag = (const float*)A;
            const long base = (long)blockIdx.z * sA + aRow0 * lda + kt;
            float4 t[4];
            #pragma unroll
            for (int r = 0; r < 4; ++r) {                    // 1024 float4 / 256 thr
                int i = tid + r * 256;
                int row = i >> 3, cg = (i & 7) * 4;
                t[r] = *(const float4*)(Ag + base + (long)row * lda + cg);
            }
            #pragma unroll
            for (int r = 0; r < 4; ++r) {
                int i = tid + r * 256;
                int o = (i >> 3) * LDSS + (i & 7) * 4;
                as[o + 0] = (bf16_t)t[r].x; as[o + 1] = (bf16_t)t[r].y;
                as[o + 2] = (bf16_t)t[r].z; as[o + 3] = (bf16_t)t[r].w;
            }
        } else {
            const bf16_t* Ag = (const bf16_t*)A;
            const long base = (long)blockIdx.z * sA + aRow0 * lda + kt;
            #pragma unroll
            for (int r = 0; r < 2; ++r) {                    // 512 x 16B / 256 thr
                int i = tid + r * 256;
                int row = i >> 2, sg = (i & 3) * 8;
                unsigned long long ga =
                    (unsigned long long)(uintptr_t)(Ag + base + (long)row * lda + sg);
                unsigned la = (unsigned)(uintptr_t)(&as[row * LDSS + sg]);
                asm volatile("global_load_async_to_lds_b128 %0, %1, off"
                             :: "v"(la), "v"(ga) : "memory");
            }
        }
        // B tile -> bs[n][k] (transposed in LDS)
        if constexpr (!BT) {
            const long base = (long)blockIdx.z * sB + (long)kt * ldb + bCol0;
            if constexpr (BDT == 0) {
                const float* Bg = (const float*)B;
                float t[8];
                #pragma unroll
                for (int r = 0; r < 8; ++r) {                // 2048 floats / 256 thr
                    int i = tid + r * 256;
                    t[r] = Bg[base + (long)(i >> 6) * ldb + (i & 63)];
                }
                #pragma unroll
                for (int r = 0; r < 8; ++r) {
                    int i = tid + r * 256;
                    bs[(i & 63) * LDSS + (i >> 6)] = (bf16_t)t[r];
                }
            } else {
                const bf16_t* Bg = (const bf16_t*)B;
                bf16_t t[8];
                #pragma unroll
                for (int r = 0; r < 8; ++r) {
                    int i = tid + r * 256;
                    t[r] = Bg[base + (long)(i >> 6) * ldb + (i & 63)];
                }
                #pragma unroll
                for (int r = 0; r < 8; ++r) {
                    int i = tid + r * 256;
                    bs[(i & 63) * LDSS + (i >> 6)] = t[r];
                }
            }
        } else {
            const long base = (long)blockIdx.z * sB + bCol0 * ldb + kt;
            if constexpr (BDT == 1) {
                const bf16_t* Bg = (const bf16_t*)B;
                {                                            // 256 x 16B / 256 thr
                    int n = tid >> 2, sg = (tid & 3) * 8;
                    unsigned long long ga =
                        (unsigned long long)(uintptr_t)(Bg + base + (long)n * ldb + sg);
                    unsigned la = (unsigned)(uintptr_t)(&bs[n * LDSS + sg]);
                    asm volatile("global_load_async_to_lds_b128 %0, %1, off"
                                 :: "v"(la), "v"(ga) : "memory");
                }
            } else {
                const float* Bg = (const float*)B;
                float4 t[2];
                #pragma unroll
                for (int r = 0; r < 2; ++r) {
                    int i = tid + r * 256;
                    t[r] = *(const float4*)(Bg + base + (long)(i >> 3) * ldb + (i & 7) * 4);
                }
                #pragma unroll
                for (int r = 0; r < 2; ++r) {
                    int i = tid + r * 256;
                    int o = (i >> 3) * LDSS + (i & 7) * 4;
                    bs[o + 0] = (bf16_t)t[r].x; bs[o + 1] = (bf16_t)t[r].y;
                    bs[o + 2] = (bf16_t)t[r].z; bs[o + 3] = (bf16_t)t[r].w;
                }
            }
        }
    };

    v8f acc[2][2] = {};
    constexpr bool kAsync = (ADT == 1) || (BT && BDT == 1);

    stage(0, 0);                                  // prologue
    int p = 0;
    for (int kt = 0; kt < K; kt += TILE_K) {
        if constexpr (kAsync)
            asm volatile("s_wait_asynccnt 0" ::: "memory");  // own copies of tile kt done
        __syncthreads();                                     // tile kt visible block-wide
        if (kt + TILE_K < K)
            stage(p ^ 1, kt + TILE_K);                       // overlap: async copy next tile

        // ---- compute: each wave does a 32x32 patch = 2x2 WMMA tiles ----
        const bf16_t* as = &As[p][0];
        const bf16_t* bs = &Bs[p][0];
        Frag fa[2], fb[2];
        #pragma unroll
        for (int s = 0; s < 2; ++s) {
            int ar = wm * 32 + s * 16 + fr;
            fa[s].h[0] = *(const v8bf*)&as[ar * LDSS + kh * 8];       // K = kh*8 .. +7
            fa[s].h[1] = *(const v8bf*)&as[ar * LDSS + 16 + kh * 8];  // K = 16+kh*8 .. +7
            int br = wn * 32 + s * 16 + fr;
            fb[s].h[0] = *(const v8bf*)&bs[br * LDSS + kh * 8];
            fb[s].h[1] = *(const v8bf*)&bs[br * LDSS + 16 + kh * 8];
        }
        #pragma unroll
        for (int i = 0; i < 2; ++i)
            #pragma unroll
            for (int j = 0; j < 2; ++j)
                acc[i][j] = __builtin_amdgcn_wmma_f32_16x16x32_bf16(
                    false, fa[i].v, false, fb[j].v, (short)0, acc[i][j], false, false);
        p ^= 1;
    }

    // ---- store D: lane holds col n=fr, rows m = v + 8*kh (v = 0..7) ----
    #pragma unroll
    for (int i = 0; i < 2; ++i) {
        #pragma unroll
        for (int j = 0; j < 2; ++j) {
            int gn = (int)bCol0 + wn * 32 + j * 16 + fr;
            long mbase = aRow0 + wm * 32 + i * 16 + 8 * kh;
            #pragma unroll
            for (int v = 0; v < 8; ++v) {
                long row = mbase + v;
                float val = acc[i][j][v];
                if constexpr (MODE == 0) {
                    val += bias[gn];
                    bf16_t bv = (bf16_t)val;
                    if (gn & 1) ((bf16_t*)C1)[row * ldc + (gn >> 1)] = bv;
                    else        ((bf16_t*)C0)[row * ldc + (gn >> 1)] = bv;
                } else if constexpr (MODE == 1) {
                    ((bf16_t*)C0)[(long)blockIdx.z * sC + row * ldc + gn] = (bf16_t)val;
                } else if constexpr (MODE == 2) {
                    ((float*)C0)[(long)blockIdx.z * sC + row * ldc + gn] = val * scale;
                } else {
                    ((float*)C0)[(long)blockIdx.z * sC + row * ldc + gn] = val + bias[gn];
                }
            }
        }
    }
}

// Per-row: rank-170 threshold (descending bitonic sort), mask to -1e-7, softmax(x/0.3).
__global__ __launch_bounds__(256)
void topk_softmax(const float* __restrict__ at, bf16_t* __restrict__ attn)
{
    __shared__ float s[1024];
    const int tid = threadIdx.x;
    const long base = ((long)blockIdx.y * 1024 + blockIdx.x) * 1024;

    float vals[4];
    #pragma unroll
    for (int r = 0; r < 4; ++r) {
        vals[r] = at[base + tid + r * 256];
        s[tid + r * 256] = vals[r];
    }
    __syncthreads();

    // bitonic sort, descending (uniform inner trip count: 4 pairs-slots per thread)
    for (int k = 2; k <= 1024; k <<= 1) {
        for (int j = k >> 1; j > 0; j >>= 1) {
            #pragma unroll
            for (int r = 0; r < 4; ++r) {
                int t = tid + r * 256;
                int ixj = t ^ j;
                if (ixj > t) {
                    float a = s[t], b = s[ixj];
                    bool keepBig = ((t & k) == 0);
                    if (keepBig ? (a < b) : (a > b)) { s[t] = b; s[ixj] = a; }
                }
            }
            __syncthreads();
        }
    }
    const float thr = s[170];                 // top_k(k=171)[-1] == rank 170 descending
    const float mn  = s[1023];
    const float mx  = s[0];
    // max of adj: masked entries become -1e-7, which can exceed mx on all-negative rows
    const float rowmax = (mn < thr) ? fmaxf(mx, -1e-7f) : mx;
    __syncthreads();

    const float invT = 1.0f / 0.3f;
    float e[4], partial = 0.0f;
    #pragma unroll
    for (int r = 0; r < 4; ++r) {
        float v = vals[r];
        float adj = (v < thr) ? -1e-7f : v;
        e[r] = __expf((adj - rowmax) * invT);
        partial += e[r];
    }
    s[tid] = partial;
    __syncthreads();
    for (int off = 128; off > 0; off >>= 1) {
        if (tid < off) s[tid] += s[tid + off];
        __syncthreads();
    }
    const float inv = 1.0f / s[0];
    #pragma unroll
    for (int r = 0; r < 4; ++r)
        attn[base + tid + r * 256] = (bf16_t)(e[r] * inv);
}

extern "C" void kernel_launch(void* const* d_in, const int* in_sizes, int n_in,
                              void* d_out, int out_size, void* d_ws, size_t ws_size,
                              hipStream_t stream) {
    const float* h      = (const float*)d_in[0];   // [8,1024,320]
    const float* Wqk    = (const float*)d_in[1];   // [320,640]
    const float* bqk    = (const float*)d_in[2];   // [640]
    const float* weight = (const float*)d_in[3];   // [320,320]
    const float* bias   = (const float*)d_in[4];   // [320]
    float* out = (float*)d_out;                    // [8,1024,320]

    char* ws = (char*)d_ws;
    size_t off = 0;
    auto carve = [&](size_t bytes) { void* p = ws + off; off += (bytes + 255) & ~(size_t)255; return p; };
    bf16_t* Qb = (bf16_t*)carve((size_t)8192 * 320 * 2);
    bf16_t* Kb = (bf16_t*)carve((size_t)8192 * 320 * 2);
    bf16_t* HW = (bf16_t*)carve((size_t)8192 * 320 * 2);
    float*  AT = (float*) carve((size_t)8 * 1024 * 1024 * 4);
    bf16_t* AP = (bf16_t*)carve((size_t)8 * 1024 * 1024 * 2);
    (void)ws_size; (void)in_sizes; (void)n_in; (void)out_size;

    dim3 blk(256);
    // K1: QK = h@Wqk + bqk, deinterleave even->Qb, odd->Kb   (M=8192, N=640, K=320)
    gemm_wmma<0, 0, false, 0><<<dim3(10, 64, 1), blk, 0, stream>>>(
        h, 320, 0, Wqk, 640, 0, Qb, Kb, 320, 0, bqk, 1.0f, 320);
    // K2: HW = h@weight                                      (M=8192, N=320, K=320)
    gemm_wmma<0, 0, false, 1><<<dim3(5, 64, 1), blk, 0, stream>>>(
        h, 320, 0, weight, 320, 0, HW, nullptr, 320, 0, nullptr, 1.0f, 320);
    // K3: AT = 0.125 * Qb @ Kb^T per batch                   (M=1024, N=1024, K=320)
    gemm_wmma<1, 1, true, 2><<<dim3(16, 8, 8), blk, 0, stream>>>(
        Qb, 320, (long)1024 * 320, Kb, 320, (long)1024 * 320,
        AT, nullptr, 1024, (long)1024 * 1024, nullptr, 0.125f, 320);
    // K4: threshold + softmax per row
    topk_softmax<<<dim3(1024, 8), blk, 0, stream>>>(AT, AP);
    // K5: out = AP @ HW + bias per batch                     (M=1024, N=320, K=1024)
    gemm_wmma<1, 1, false, 3><<<dim3(5, 8, 8), blk, 0, stream>>>(
        AP, 1024, (long)1024 * 1024, HW, 320, (long)1024 * 320,
        out, nullptr, 320, (long)1024 * 320, bias, 1.0f, 1024);
}
